// DMN_19404662243510
// MI455X (gfx1250) — matrix operand order repaired and checked by reading, not compiled, not run on hardware
//
#include <hip/hip_runtime.h>
#include <hip/hip_fp16.h>
#include <math.h>

// ---------------------------------------------------------------- problem dims
constexpr int kB = 32, kS = 1024, kSQ = 32, kN = 128, kDIN = 128;
constexpr int kH = 256, kA = 100, kO = 2000, kG3 = 768;     // 3*H
constexpr int kFW = 1824;   // feats width 7H+2=1794 padded to mult of 32
constexpr int kXAW = 2272;  // O+H=2256 padded to mult of 32
constexpr int kH1W = 112;   // A=100 padded to mult of 16

typedef __attribute__((ext_vector_type(16))) _Float16 v16h;
typedef __attribute__((ext_vector_type(8)))  float    v8f;

// ---------------------------------------------------------------- WMMA helpers
__device__ inline v8f wmma16(v16h a, v16h b, v8f c) {
  // v_wmma_f32_16x16x32_f16 : D = A(16x32 f16) * B(32x16 f16) + C(16x16 f32)
  return __builtin_amdgcn_wmma_f32_16x16x32_f16(false, a, false, b, (short)0, c,
                                                false, false);
}

// A fragment (16x32 f16) gathered from f32 row-major (ISA 7.12.2):
// lane L: m = L&15, half = L>>4 ; element j: v=j>>1,
//   k = (v>=4 ? 16:0) + (v&3)*2 + half*8 + (j&1)
__device__ inline v16h load_a_f32(const float* __restrict__ A, int lda, int m0, int k0) {
  const int lane = threadIdx.x & 31;
  const float* row = A + (size_t)(m0 + (lane & 15)) * lda + k0 + ((lane >> 4) << 3);
  v16h a;
#pragma unroll
  for (int j = 0; j < 16; ++j) {
    const int v = j >> 1;
    const int k = ((v & 4) << 2) + ((v & 3) << 1) + (j & 1);
    a[j] = (_Float16)row[k];
  }
  return a;
}

// B fragment (32x16 f16), pre-packed: lane L = col N=L&15, element j -> K=(j&8)*2+(L>>4)*8+(j&7).
__device__ inline v16h load_b_frag(const _Float16* __restrict__ Bp, size_t fragIdx) {
  const int lane = threadIdx.x & 31;
  return *(const v16h*)(Bp + (fragIdx * 32 + lane) * 16);
}

// C/D fragment store: VGPR r -> M = r + 8*(lane>>4), N = lane&15
__device__ inline void store_c_bias(float* __restrict__ C, int ldc, int m0, int n0,
                                    v8f acc, const float* __restrict__ bias, int Nreal) {
  const int lane = threadIdx.x & 31;
  const int n = n0 + (lane & 15);
  const int mb = m0 + ((lane >> 4) << 3);
  const float bv = (bias != nullptr && n < Nreal) ? bias[n] : 0.f;
#pragma unroll
  for (int r = 0; r < 8; ++r) C[(size_t)(mb + r) * ldc + n] = acc[r] + bv;
}

// ---------------------------------------------------------------- weight packer
// B-fragment order. transp=1: B[k][n] = W[n][k] (the "@ w.T" case)
__global__ void k_pack(const float* __restrict__ W, _Float16* __restrict__ Bp,
                       int Wcols, int Kreal, int Nreal, int Ktiles, int Ntiles,
                       int transp) {
  const size_t total = (size_t)Ktiles * Ntiles * 512;
  for (size_t idx = (size_t)blockIdx.x * blockDim.x + threadIdx.x; idx < total;
       idx += (size_t)gridDim.x * blockDim.x) {
    const int f = (int)(idx >> 9);
    const int rem = (int)(idx & 511);
    const int lane = rem >> 4, j = rem & 15;
    const int kt = f / Ntiles, nt = f % Ntiles;
    const int k = kt * 32 + ((j & 8) << 1) + ((lane >> 4) << 3) + (j & 7), n = nt * 16 + (lane & 15);
    float v = 0.f;
    if (k < Kreal && n < Nreal)
      v = transp ? W[(size_t)n * Wcols + k] : W[(size_t)k * Wcols + n];
    Bp[idx] = (_Float16)v;
  }
}

// Pack activations (B,S,D) f32 -> time-major A-fragments [Mt][Kt][32][16] f16.
// Row m of the logical GEMM = s*32 + b  (time-major), folding the transpose in.
__global__ void k_packA(const float* __restrict__ src, _Float16* __restrict__ Ap,
                        int Ss, int Dd, int Ktiles) {
  const size_t total = (size_t)(Ss * kB / 16) * Ktiles * 512;
  for (size_t idx = (size_t)blockIdx.x * blockDim.x + threadIdx.x; idx < total;
       idx += (size_t)gridDim.x * blockDim.x) {
    const int f = (int)(idx >> 9);
    const int rem = (int)(idx & 511);
    const int ln = rem >> 4, j = rem & 15;
    const int mt = f / Ktiles, kt = f % Ktiles;
    const int m = mt * 16 + (ln & 15);
    const int v = j >> 1;
    const int k = kt * 32 + ((v & 4) << 2) + ((v & 3) << 1) + ((ln >> 4) << 3) + (j & 1);
    const int s = m >> 5, b = m & 31;  // kB == 32
    Ap[idx] = (_Float16)src[((size_t)b * Ss + s) * Dd + k];
  }
}

// ---------------------------------------------------------------- generic GEMM
// C[M,N] = A[M,K](f32) * Bpacked + bias. One 16x16 tile per wave (odd shapes).
__global__ void k_gemm(const float* __restrict__ A, int lda,
                       const _Float16* __restrict__ Bp,
                       float* __restrict__ C, int ldc,
                       const float* __restrict__ bias, int Nreal,
                       int Mtiles, int Ntiles, int Ktiles) {
  const int wave = blockIdx.x * (blockDim.x >> 5) + (threadIdx.x >> 5);
  if (wave >= Mtiles * Ntiles) return;  // wave-uniform
  const int mt = wave / Ntiles;
  const int nt = wave % Ntiles;
  v8f acc = {};
  for (int kt = 0; kt < Ktiles; ++kt) {
    if (kt + 1 < Ktiles)  // global_prefetch_b8 of next B fragment
      __builtin_prefetch(Bp + ((size_t)(kt + 1) * Ntiles + nt) * 512, 0, 1);
    v16h a = load_a_f32(A, lda, mt * 16, kt * 32);
    v16h b = load_b_frag(Bp, (size_t)kt * Ntiles + nt);
    acc = wmma16(a, b, acc);
  }
  store_c_bias(C, ldc, mt * 16, nt * 16, acc, bias, Nreal);
}

// ----------------------------------------------------- blocked async-LDS GEMM
// Packed-A variant: each wave owns 4 N-tiles; B fragments are double-buffered
// through wave-private LDS with global_load_async_to_lds_b128 (ASYNCcnt),
// prefetching kt+1 while WMMA consumes kt. One 32B A load feeds 4 WMMAs.
__global__ __launch_bounds__(256, 1)
void k_gemm4p(const _Float16* __restrict__ Ap, const _Float16* __restrict__ Bp,
              float* __restrict__ C, int ldc, const float* __restrict__ bias,
              int Nreal, int Mtiles, int Ntiles, int Ktiles) {
  __shared__ __attribute__((aligned(32))) _Float16 dbuf[8][2][4][512];  // 64 KB
  const int wv = threadIdx.x >> 5, lane = threadIdx.x & 31;
  const int nGroups = Ntiles >> 2;  // Ntiles % 4 == 0
  const int wave = blockIdx.x * 8 + wv;
  if (wave >= Mtiles * nGroups) return;  // wave-uniform
  const int mt = wave / nGroups;
  const int nt0 = (wave % nGroups) << 2;

  auto issue = [&](int kt, int buf) {
    // 4 fragments * 32B/lane -> 8 async B128 loads per lane
#pragma unroll
    for (int ff = 0; ff < 4; ++ff) {
      const _Float16* src = Bp + ((size_t)(kt * Ntiles + nt0 + ff) * 32 + lane) * 16;
      const unsigned dst = (unsigned)(size_t)&dbuf[wv][buf][ff][lane * 16];
      asm volatile("global_load_async_to_lds_b128 %0, %1, off\n\t"
                   "global_load_async_to_lds_b128 %0, %1, off offset:16"
                   :: "v"(dst), "v"(src) : "memory");
    }
  };

  v8f acc0 = {}, acc1 = {}, acc2 = {}, acc3 = {};
  issue(0, 0);
  for (int kt = 0; kt < Ktiles; ++kt) {
    const int buf = kt & 1;
    if (kt + 1 < Ktiles) {
      issue(kt + 1, buf ^ 1);
      // 8 newer loads outstanding; <=8 means current buffer landed (in-order)
      asm volatile("s_wait_asynccnt 0x8" ::: "memory");
    } else {
      asm volatile("s_wait_asynccnt 0x0" ::: "memory");
    }
    v16h a = *(const v16h*)(Ap + ((size_t)(mt * Ktiles + kt) * 32 + lane) * 16);
    acc0 = wmma16(a, *(const v16h*)&dbuf[wv][buf][0][lane * 16], acc0);
    acc1 = wmma16(a, *(const v16h*)&dbuf[wv][buf][1][lane * 16], acc1);
    acc2 = wmma16(a, *(const v16h*)&dbuf[wv][buf][2][lane * 16], acc2);
    acc3 = wmma16(a, *(const v16h*)&dbuf[wv][buf][3][lane * 16], acc3);
  }
  store_c_bias(C, ldc, mt * 16, (nt0 + 0) * 16, acc0, bias, Nreal);
  store_c_bias(C, ldc, mt * 16, (nt0 + 1) * 16, acc1, bias, Nreal);
  store_c_bias(C, ldc, mt * 16, (nt0 + 2) * 16, acc2, bias, Nreal);
  store_c_bias(C, ldc, mt * 16, (nt0 + 3) * 16, acc3, bias, Nreal);
}

// ---------------------------------------------------------------- GRU recurrence
// h (32x256) lives in LDS as f32 plus a pre-packed WMMA A-fragment image.
// 32 waves: wave w -> mt=w&1, ntr=w>>1 owns gate triplet (ntr, ntr+16, ntr+32),
// so the GRU pointwise is wave-local and each thread scatters its new h values
// straight into the A-fragment image (k index of h == n): 2 barriers/step.
__device__ inline void rebuild_afrag(_Float16 (*aFrag)[32][16],
                                     const float (*hbuf)[kH], int tid) {
  for (int e = tid; e < 16 * 512; e += 1024) {
    const int f = e >> 9, rem = e & 511;
    const int ln = rem >> 4, j = rem & 15;
    const int ktA = f >> 1, mtA = f & 1;
    const int m = mtA * 16 + (ln & 15);
    const int v = j >> 1;
    const int k = ((v & 4) << 2) + ((v & 3) << 1) + ((ln >> 4) << 3) + (j & 1);
    aFrag[f][ln][j] = (_Float16)hbuf[m][ktA * 32 + k];
  }
}

__global__ __launch_bounds__(1024, 1)
void k_gru_scan(const float* __restrict__ gi,       // [T][32][768] time-major
                const _Float16* __restrict__ whhP,  // packed, Ntiles=48,Ktiles=8
                const float* __restrict__ bhh,      // [768]
                const float* __restrict__ h0,       // [32*256]
                float* __restrict__ outAll,         // [T][32][256] or null
                float* __restrict__ hT,             // [32*256] or null
                int T) {
  __shared__ __attribute__((aligned(32))) _Float16 aFrag[16][32][16];  // 16 KB
  __shared__ float hbuf[kB][kH];                                       // 32 KB
  const int tid = threadIdx.x;
  const int wave = tid >> 5, lane = tid & 31;
  for (int i = tid; i < kB * kH; i += 1024) hbuf[i >> 8][i & 255] = h0[i];
  __syncthreads();
  rebuild_afrag(aFrag, hbuf, tid);
  __syncthreads();

  const int mt = wave & 1, ntr = wave >> 1;
  const int n = ntr * 16 + (lane & 15);
  const int mb = mt * 16 + ((lane >> 4) << 3);
  // where h[.][n] lives inside the A-fragment image (k == n):
  const int kl = n & 31;
  const int vv = ((kl >> 4) & 1) * 4 + ((kl >> 1) & 3);
  const int jj = vv * 2 + (kl & 1);
  const int halfK = (kl >> 3) & 1;
  const int fA = (n >> 5) * 2 + mt;

  for (int t = 0; t < T; ++t) {
    v8f aR = {}, aZ = {}, aN = {};
#pragma unroll
    for (int kt = 0; kt < 8; ++kt) {
      v16h a = *(const v16h*)&aFrag[kt * 2 + mt][lane][0];
      aR = wmma16(a, load_b_frag(whhP, (size_t)kt * 48 + ntr), aR);
      aZ = wmma16(a, load_b_frag(whhP, (size_t)kt * 48 + ntr + 16), aZ);
      aN = wmma16(a, load_b_frag(whhP, (size_t)kt * 48 + ntr + 32), aN);
    }
    const float* giT = gi + (size_t)t * kB * kG3;
    float hnew[8];
#pragma unroll
    for (int r = 0; r < 8; ++r) {
      const int m = mb + r;
      const float rg = 1.f / (1.f + expf(-(giT[m * kG3 + n] + aR[r] + bhh[n])));
      const float zg = 1.f / (1.f + expf(-(giT[m * kG3 + n + 256] + aZ[r] + bhh[n + 256])));
      const float nn = tanhf(giT[m * kG3 + n + 512] + rg * (aN[r] + bhh[n + 512]));
      hnew[r] = (1.f - zg) * nn + zg * hbuf[m][n];
    }
    __syncthreads();  // all reads of hbuf/aFrag for step t complete
#pragma unroll
    for (int r = 0; r < 8; ++r) {
      const int m = mb + r;
      hbuf[m][n] = hnew[r];
      aFrag[fA][(m & 15) + halfK * 16][jj] = (_Float16)hnew[r];
      if (outAll) outAll[((size_t)t * kB + m) * kH + n] = hnew[r];
    }
    __syncthreads();  // new h image visible to all waves
  }
  if (hT)
    for (int i = tid; i < kB * kH; i += 1024) hT[i] = hbuf[i >> 8][i & 255];
}

// ---------------------------------------------------------------- small kernels
__global__ void k_gather(const float* __restrict__ outAll, const int* __restrict__ cidx,
                         float* __restrict__ c_sel) {
  const int total = kB * kN * kH;
  for (int i = blockIdx.x * blockDim.x + threadIdx.x; i < total;
       i += gridDim.x * blockDim.x) {
    const int h = i & 255, bi = i >> 8;
    const int b = bi >> 7, ii = bi & 127;
    const int t = cidx[b * kN + ii];
    c_sel[i] = outAll[((size_t)t * kB + b) * kH + h];
  }
}

__global__ void k_feats(const float* __restrict__ c_sel, const float* __restrict__ cw,
                        const float* __restrict__ qh, const float* __restrict__ m,
                        float* __restrict__ feats) {
  __shared__ float red[256];
  __shared__ float dots[2];
  const int bi = blockIdx.x;  // b*128+i
  const int b = bi >> 7;
  const int tid = threadIdx.x;  // 256 == H
  const float cv = c_sel[(size_t)bi * kH + tid];
  const float qv = qh[b * kH + tid];
  const float mv = m[b * kH + tid];
  const float cwv = cw[(size_t)bi * kH + tid];
  red[tid] = cwv * qv; __syncthreads();
  for (int s = 128; s > 0; s >>= 1) { if (tid < s) red[tid] += red[tid + s]; __syncthreads(); }
  if (tid == 0) dots[0] = red[0];
  __syncthreads();
  red[tid] = cwv * mv; __syncthreads();
  for (int s = 128; s > 0; s >>= 1) { if (tid < s) red[tid] += red[tid + s]; __syncthreads(); }
  if (tid == 0) dots[1] = red[0];
  __syncthreads();
  float* fr = feats + (size_t)bi * kFW;
  fr[tid] = cv;
  fr[kH + tid] = mv;
  fr[2 * kH + tid] = qv;
  fr[3 * kH + tid] = cv * qv;
  fr[4 * kH + tid] = cv * mv;
  fr[5 * kH + tid] = fabsf(cv - qv);
  fr[6 * kH + tid] = fabsf(cv - mv);
  if (tid == 0) { fr[7 * kH] = dots[0]; fr[7 * kH + 1] = dots[1]; }
  if (tid >= 2 && 7 * kH + tid < kFW) fr[7 * kH + tid] = 0.f;  // zero K-padding
}

__global__ void k_score(const float* __restrict__ h1pre, const float* __restrict__ w2,
                        const float* __restrict__ b2, const int* __restrict__ len_c,
                        const float* __restrict__ c_sel, float* __restrict__ p_out,
                        float* __restrict__ e) {
  __shared__ float sc[kN];
  __shared__ float red[kN];
  const int b = blockIdx.x;
  const int tid = threadIdx.x;  // 128 == N
  const int len = len_c[b];
  {
    float s = b2[0];
    const float* hr = h1pre + (size_t)(b * kN + tid) * kH1W;
    for (int j = 0; j < kA; ++j) s += w2[j] * tanhf(hr[j]);
    sc[tid] = s;
  }
  __syncthreads();
  red[tid] = (tid < len) ? sc[tid] : -INFINITY; __syncthreads();
  for (int s2 = 64; s2 > 0; s2 >>= 1) { if (tid < s2) red[tid] = fmaxf(red[tid], red[tid + s2]); __syncthreads(); }
  const float mx = red[0];
  __syncthreads();
  const float ex = (tid < len) ? expf(sc[tid] - mx) : 0.f;
  red[tid] = ex; __syncthreads();
  for (int s2 = 64; s2 > 0; s2 >>= 1) { if (tid < s2) red[tid] += red[tid + s2]; __syncthreads(); }
  const float p = ex / red[0];
  __syncthreads();
  sc[tid] = p;
  p_out[b * kN + tid] = p;
  __syncthreads();
  for (int h = tid; h < kH; h += kN) {
    float acc = 0.f;
    for (int i = 0; i < kN; ++i) acc += sc[i] * c_sel[((size_t)b * kN + i) * kH + h];
    e[b * kH + h] = acc;
  }
}

__global__ void k_gru_pw(const float* __restrict__ gi, const float* __restrict__ gh,
                         const float* __restrict__ hOld, float* __restrict__ hNew) {
  const int i = blockIdx.x * blockDim.x + threadIdx.x;
  if (i >= kB * kH) return;
  const int b = i >> 8, n = i & 255;
  const float* gib = gi + b * kG3;
  const float* ghb = gh + b * kG3;
  const float r = 1.f / (1.f + expf(-(gib[n] + ghb[n])));
  const float z = 1.f / (1.f + expf(-(gib[n + 256] + ghb[n + 256])));
  const float nn = tanhf(gib[n + 512] + r * ghb[n + 512]);
  hNew[i] = (1.f - z) * nn + z * hOld[i];
}

__global__ void k_softmax(const float* __restrict__ X, float* __restrict__ Y, int cols) {
  __shared__ float red[256];
  const int row = blockIdx.x;
  const int tid = threadIdx.x;
  const float* x = X + (size_t)row * cols;
  float mx = -INFINITY;
  for (int c = tid; c < cols; c += 256) mx = fmaxf(mx, x[c]);
  red[tid] = mx; __syncthreads();
  for (int s = 128; s > 0; s >>= 1) { if (tid < s) red[tid] = fmaxf(red[tid], red[tid + s]); __syncthreads(); }
  mx = red[0]; __syncthreads();
  float sum = 0.f;
  for (int c = tid; c < cols; c += 256) sum += expf(x[c] - mx);
  red[tid] = sum; __syncthreads();
  for (int s = 128; s > 0; s >>= 1) { if (tid < s) red[tid] += red[tid + s]; __syncthreads(); }
  sum = red[0];
  float* y = Y + (size_t)row * cols;
  for (int c = tid; c < cols; c += 256) y[c] = expf(x[c] - mx) / sum;
}

__global__ void k_copy(const float* __restrict__ s, float* __restrict__ d, int n) {
  const int i = blockIdx.x * blockDim.x + threadIdx.x;
  if (i < n) d[i] = s[i];
}

__global__ void k_build_xa(const float* __restrict__ y, const float* __restrict__ qh,
                           float* __restrict__ xa) {
  const int total = kB * kXAW;
  for (int i = blockIdx.x * blockDim.x + threadIdx.x; i < total;
       i += gridDim.x * blockDim.x) {
    const int b = i / kXAW, c = i % kXAW;
    float v;
    if (c < kO) v = y[b * kO + c];
    else if (c < kO + kH) v = qh[b * kH + (c - kO)];
    else v = 0.f;  // zero K-padding
    xa[i] = v;
  }
}

// ---------------------------------------------------------------- host driver
extern "C" void kernel_launch(void* const* d_in, const int* in_sizes, int n_in,
                              void* d_out, int out_size, void* d_ws, size_t ws_size,
                              hipStream_t stream) {
  (void)in_sizes; (void)n_in; (void)out_size; (void)ws_size;
  const float* c        = (const float*)d_in[0];
  const float* q        = (const float*)d_in[1];
  const float* i_state  = (const float*)d_in[2];
  const float* q_state  = (const float*)d_in[3];
  const float* in_w_ih  = (const float*)d_in[4];
  const float* in_w_hh  = (const float*)d_in[5];
  const float* in_b_ih  = (const float*)d_in[6];
  const float* in_b_hh  = (const float*)d_in[7];
  const float* qe_w_ih  = (const float*)d_in[8];
  const float* qe_w_hh  = (const float*)d_in[9];
  const float* qe_b_ih  = (const float*)d_in[10];
  const float* qe_b_hh  = (const float*)d_in[11];
  const float* att_weight = (const float*)d_in[12];
  const float* att_w1   = (const float*)d_in[13];
  const float* att_b1   = (const float*)d_in[14];
  const float* att_w2   = (const float*)d_in[15];
  const float* att_b2   = (const float*)d_in[16];
  const float* mem_w_ih = (const float*)d_in[17];
  const float* mem_w_hh = (const float*)d_in[18];
  const float* mem_b_ih = (const float*)d_in[19];
  const float* mem_b_hh = (const float*)d_in[20];
  const float* out_w    = (const float*)d_in[21];
  const float* out_b    = (const float*)d_in[22];
  const float* ans_w_ih = (const float*)d_in[23];
  const float* ans_w_hh = (const float*)d_in[24];
  const float* ans_b_ih = (const float*)d_in[25];
  const float* ans_b_hh = (const float*)d_in[26];
  const int*   c_index  = (const int*)d_in[27];
  const int*   len_c    = (const int*)d_in[28];
  float* outF = (float*)d_out;  // [ y(32*2000) | att(3*32*128) ]

  // ---- workspace carve ----
  char* base = (char*)d_ws;
  size_t off = 0;
  auto carve = [&](size_t bytes) -> char* {
    char* p = base + off;
    off = (off + bytes + 255) & ~(size_t)255;
    return p;
  };
  auto cf16 = [&](size_t n) { return (_Float16*)carve(n * sizeof(_Float16)); };
  auto cf32 = [&](size_t n) { return (float*)carve(n * sizeof(float)); };

  _Float16* p_in_ih  = cf16((size_t)4 * 48 * 512);
  _Float16* p_in_hh  = cf16((size_t)8 * 48 * 512);
  _Float16* p_qe_ih  = cf16((size_t)4 * 48 * 512);
  _Float16* p_qe_hh  = cf16((size_t)8 * 48 * 512);
  _Float16* p_att    = cf16((size_t)8 * 16 * 512);
  _Float16* p_w1     = cf16((size_t)57 * 7 * 512);
  _Float16* p_mem_ih = cf16((size_t)8 * 48 * 512);
  _Float16* p_mem_hh = cf16((size_t)8 * 48 * 512);
  _Float16* p_outw   = cf16((size_t)8 * 125 * 512);
  _Float16* p_ans_ih = cf16((size_t)71 * 48 * 512);
  _Float16* p_ans_hh = cf16((size_t)8 * 48 * 512);
  _Float16* aPack_c  = cf16((size_t)2048 * 4 * 512);  // c as A-fragments
  _Float16* aPack_q  = cf16((size_t)64 * 4 * 512);    // q as A-fragments

  float* gi_c   = cf32((size_t)kS * kB * kG3);
  float* gi_q   = cf32((size_t)kSQ * kB * kG3);
  float* c_out  = cf32((size_t)kS * kB * kH);
  float* q_hT   = cf32((size_t)kB * kH);
  float* c_sel  = cf32((size_t)kB * kN * kH);
  float* cw     = cf32((size_t)kB * kN * kH);
  float* feats  = cf32((size_t)kB * kN * kFW);
  float* h1pre  = cf32((size_t)kB * kN * kH1W);
  float* evec   = cf32((size_t)kB * kH);
  float* mA     = cf32((size_t)kB * kH);
  float* mBuf2  = cf32((size_t)kB * kH);
  float* msqA   = cf32((size_t)kB * kH);
  float* msqB   = cf32((size_t)kB * kH);
  float* gis    = cf32((size_t)kB * kG3);
  float* ghs    = cf32((size_t)kB * kG3);
  float* logits = cf32((size_t)kB * kO);
  float* ybuf   = cf32((size_t)kB * kO);
  float* xa     = cf32((size_t)kB * kXAW);

  const dim3 blk(256);
  auto g1 = [](size_t n) { return dim3((unsigned)((n + 255) / 256)); };
  auto gG = [](int Mt, int Nt) { return dim3((unsigned)((Mt * Nt + 7) / 8)); };
  auto gG4 = [](int Mt, int Nt) { return dim3((unsigned)((Mt * (Nt >> 2) + 7) / 8)); };

  // ---- pack weights into WMMA B-fragment layout (f16) ----
  k_pack<<<g1(4u*48*512), blk, 0, stream>>>(in_w_ih,  p_in_ih,  kDIN, kDIN, kG3, 4, 48, 1);
  k_pack<<<g1(8u*48*512), blk, 0, stream>>>(in_w_hh,  p_in_hh,  kH,   kH,   kG3, 8, 48, 1);
  k_pack<<<g1(4u*48*512), blk, 0, stream>>>(qe_w_ih,  p_qe_ih,  kDIN, kDIN, kG3, 4, 48, 1);
  k_pack<<<g1(8u*48*512), blk, 0, stream>>>(qe_w_hh,  p_qe_hh,  kH,   kH,   kG3, 8, 48, 1);
  k_pack<<<g1(8u*16*512), blk, 0, stream>>>(att_weight, p_att,  kH,   kH,   kH,  8, 16, 0);
  k_pack<<<g1(57u*7*512), blk, 0, stream>>>(att_w1,   p_w1,     1794, 1794, kA, 57, 7, 1);
  k_pack<<<g1(8u*48*512), blk, 0, stream>>>(mem_w_ih, p_mem_ih, kH,   kH,   kG3, 8, 48, 1);
  k_pack<<<g1(8u*48*512), blk, 0, stream>>>(mem_w_hh, p_mem_hh, kH,   kH,   kG3, 8, 48, 1);
  k_pack<<<g1(8u*125*512), blk, 0, stream>>>(out_w,   p_outw,   kH,   kH,   kO,  8, 125, 1);
  k_pack<<<g1(71u*48*512), blk, 0, stream>>>(ans_w_ih, p_ans_ih, 2256, 2256, kG3, 71, 48, 1);
  k_pack<<<g1(8u*48*512), blk, 0, stream>>>(ans_w_hh, p_ans_hh, kH,   kH,   kG3, 8, 48, 1);

  // ---- pack activations as A-fragments (folds (B,S,D)->time-major) ----
  k_packA<<<g1((size_t)2048*4*512), blk, 0, stream>>>(c, aPack_c, kS,  kDIN, 4);
  k_packA<<<g1((size_t)64*4*512),  blk, 0, stream>>>(q, aPack_q, kSQ, kDIN, 4);

  // ---- batched input projections gi = x @ w_ih.T + b_ih (async-LDS WMMA) ----
  k_gemm4p<<<gG4(2048, 48), blk, 0, stream>>>(aPack_c, p_in_ih, gi_c, kG3, in_b_ih, kG3, 2048, 48, 4);
  k_gemm4p<<<gG4(64, 48),  blk, 0, stream>>>(aPack_q, p_qe_ih, gi_q, kG3, qe_b_ih, kG3, 64, 48, 4);

  // ---- sequential GRU scans (persistent single workgroup, h in LDS frags) ----
  k_gru_scan<<<1, 1024, 0, stream>>>(gi_c, p_in_hh, in_b_hh, i_state, c_out, nullptr, kS);
  k_gru_scan<<<1, 1024, 0, stream>>>(gi_q, p_qe_hh, qe_b_hh, q_state, nullptr, q_hT, kSQ);

  // ---- attention setup ----
  k_gather<<<g1((size_t)kB*kN*kH), blk, 0, stream>>>(c_out, c_index, c_sel);
  k_gemm<<<gG(256, 16), blk, 0, stream>>>(c_sel, kH, p_att, cw, kH, nullptr, kH, 256, 16, 8);
  k_copy<<<g1(kB*kH), blk, 0, stream>>>(q_hT, mA, kB * kH);

  // ---- episodic memory loop ----
  float* mCur = mA; float* mNxt = mBuf2;
  for (int it = 0; it < 3; ++it) {
    k_feats<<<dim3(kB * kN), dim3(256), 0, stream>>>(c_sel, cw, q_hT, mCur, feats);
    k_gemm<<<gG(256, 7), blk, 0, stream>>>(feats, kFW, p_w1, h1pre, kH1W, att_b1, kA, 256, 7, 57);
    k_score<<<dim3(kB), dim3(kN), 0, stream>>>(h1pre, att_w2, att_b2, len_c, c_sel,
                                               outF + kB * kO + it * kB * kN, evec);
    k_gemm<<<gG(2, 48), blk, 0, stream>>>(evec, kH, p_mem_ih, gis, kG3, mem_b_ih, kG3, 2, 48, 8);
    k_gemm<<<gG(2, 48), blk, 0, stream>>>(mCur, kH, p_mem_hh, ghs, kG3, mem_b_hh, kG3, 2, 48, 8);
    k_gru_pw<<<g1(kB*kH), blk, 0, stream>>>(gis, ghs, mCur, mNxt);
    float* t = mCur; mCur = mNxt; mNxt = t;
  }

  // ---- answer module ----
  k_copy<<<g1(kB*kH), blk, 0, stream>>>(mCur, msqA, kB * kH);
  float* sCur = msqA; float* sNxt = msqB;
  for (int k = 0; k < 2; ++k) {
    k_gemm<<<gG(2, 125), blk, 0, stream>>>(sCur, kH, p_outw, logits, kO, out_b, kO, 2, 125, 8);
    k_softmax<<<dim3(kB), dim3(256), 0, stream>>>(logits, ybuf, kO);
    k_build_xa<<<g1(kB*kXAW), blk, 0, stream>>>(ybuf, q_hT, xa);
    k_gemm<<<gG(2, 48), blk, 0, stream>>>(xa, kXAW, p_ans_ih, gis, kG3, ans_b_ih, kG3, 2, 48, 71);
    k_gemm<<<gG(2, 48), blk, 0, stream>>>(sCur, kH, p_ans_hh, ghs, kG3, ans_b_hh, kG3, 2, 48, 8);
    k_gru_pw<<<g1(kB*kH), blk, 0, stream>>>(gis, ghs, sCur, sNxt);
    float* t = sCur; sCur = sNxt; sNxt = t;
  }
  k_gemm<<<gG(2, 125), blk, 0, stream>>>(sCur, kH, p_outw, logits, kO, out_b, kO, 2, 125, 8);
  k_softmax<<<dim3(kB), dim3(256), 0, stream>>>(logits, outF, kO);
}
